// MovingAverageGatedAttention_55886114456153
// MI455X (gfx1250) — compile-verified, matrix operand order checked by reading, not run on hardware
//
#include <hip/hip_runtime.h>
#include <hip/hip_bf16.h>

#define D_    1024
#define Z_    128
#define H_    2048
#define NEMA  16
#define C_    256
#define L_    4096
#define B_    4
#define ROWS_ (L_*B_)              // 16384
#define MXN_  (Z_ + H_ + 2*D_)     // 4224

typedef __attribute__((ext_vector_type(16))) _Float16 v16h;
typedef __attribute__((ext_vector_type(8)))  float    v8f;

// Fast sigmoid/silu: v_exp_f32 + v_rcp_f32 (TRANS ops, co-execute with VALU)
// instead of the IEEE divide expansion.
__device__ __forceinline__ float sigm_(float x){ return __builtin_amdgcn_rcpf(1.0f+__expf(-x)); }
__device__ __forceinline__ float silu_(float x){ return x * sigm_(x); }

union FragU { uint4 u[2]; v16h v; };

// Load one 16x32 f16 fragment (A layout; B uses same per-lane layout with N as "row")
// from LDS tile stored row-major [128][40] halfs (80B row stride -> 16B aligned).
__device__ __forceinline__ v16h ld_frag(const _Float16 (*s)[40], int row0, int lane){
  int r  = row0 + (lane & 15);
  int kh = (lane >> 4) * 8;                 // lanes 0-15: K 0-7/16-23, lanes 16-31: K 8-15/24-31
  FragU f;
  f.u[0] = *(const uint4*)(&s[r][kh]);      // halfs kh..kh+7
  f.u[1] = *(const uint4*)(&s[r][kh + 16]); // halfs kh+16..kh+23
  return f.v;
}

// 128x128 block tile, 256 threads = 8 waves (4x2), each wave 32x64 = 2x4 WMMA tiles.
// B_NK: global B stored N-major (rows of length K); else K-major (K x N, ldb = N).
template<bool B_NK>
__device__ __forceinline__ void gemm_tile(
    const _Float16* __restrict__ A, long lda,
    const _Float16* __restrict__ B, long ldb,
    int K, int blockM, int blockN,
    _Float16 (*sA)[40], _Float16 (*sB)[40],
    v8f acc[2][4])
{
  const int t    = threadIdx.x;
  const int lane = t & 31;
  const int wave = t >> 5;
  const int wm   = (wave >> 1) * 32;
  const int wn   = (wave & 1) * 64;

  for (int k0 = 0; k0 < K; k0 += 32) {
    { // stage A: 128 rows x 32 halfs, 2 threads/row, 16 halfs each
      int row = t >> 1, kof = (t & 1) * 16;
      const _Float16* g = A + (long)(blockM + row) * lda + k0 + kof;
      uint4 d0 = *(const uint4*)g;
      uint4 d1 = *(const uint4*)(g + 8);
      *(uint4*)(&sA[row][kof])     = d0;
      *(uint4*)(&sA[row][kof + 8]) = d1;
    }
    if (B_NK) {
      int row = t >> 1, kof = (t & 1) * 16;
      const _Float16* g = B + (long)(blockN + row) * ldb + k0 + kof;
      uint4 d0 = *(const uint4*)g;
      uint4 d1 = *(const uint4*)(g + 8);
      *(uint4*)(&sB[row][kof])     = d0;
      *(uint4*)(&sB[row][kof + 8]) = d1;
    } else { // transpose K x N -> N x K into LDS
      int n  = t & 127;
      int kk = (t >> 7) * 16;
      #pragma unroll
      for (int i = 0; i < 16; ++i)
        sB[n][kk + i] = B[(long)(k0 + kk + i) * ldb + blockN + n];
    }
    __syncthreads();

    v16h af0 = ld_frag(sA, wm,      lane);
    v16h af1 = ld_frag(sA, wm + 16, lane);
    v16h bf[4];
    #pragma unroll
    for (int j = 0; j < 4; ++j) bf[j] = ld_frag(sB, wn + j*16, lane);

    #pragma unroll
    for (int j = 0; j < 4; ++j)
      acc[0][j] = __builtin_amdgcn_wmma_f32_16x16x32_f16(false, af0, false, bf[j],
                                                         (short)0, acc[0][j], false, false);
    #pragma unroll
    for (int j = 0; j < 4; ++j)
      acc[1][j] = __builtin_amdgcn_wmma_f32_16x16x32_f16(false, af1, false, bf[j],
                                                         (short)0, acc[1][j], false, false);
    __syncthreads();
  }
}

#define EPILOGUE_BEGIN                                            \
  const int lane = threadIdx.x & 31;                              \
  const int wave = threadIdx.x >> 5;                              \
  const int wm   = (wave >> 1) * 32;                              \
  const int wn   = (wave & 1) * 64;                               \
  _Pragma("unroll") for (int i = 0; i < 2; ++i)                   \
  _Pragma("unroll") for (int j = 0; j < 4; ++j)                   \
  _Pragma("unroll") for (int e = 0; e < 8; ++e) {                 \
    int r = bm + wm + i*16 + ((lane >> 4) << 3) + e;              \
    int c = bn + wn + j*16 + (lane & 15);                         \
    float av = acc[i][j][e];
#define EPILOGUE_END }

// ---------- GEMM 1: v = silu(xn @ v_w + v_b) ----------
__global__ void __launch_bounds__(256)
k_gemm_v(const _Float16* __restrict__ xnh, const _Float16* __restrict__ vw16,
         const float* __restrict__ v_b, _Float16* __restrict__ vbuf) {
  __shared__ _Float16 sA[128][40];
  __shared__ _Float16 sB[128][40];
  int bm = blockIdx.y * 128, bn = blockIdx.x * 128;
  v8f acc[2][4] = {};
  gemm_tile<false>(xnh, D_, vw16, H_, D_, bm, bn, sA, sB, acc);
  EPILOGUE_BEGIN
    vbuf[(long)r * H_ + c] = (_Float16)silu_(av + v_b[c]);
  EPILOGUE_END
}

// ---------- GEMM 2: base = mx @ mx_w + mx_b, split epilogue ----------
__global__ void __launch_bounds__(256)
k_gemm_base(const _Float16* __restrict__ mxh, const _Float16* __restrict__ mxw16,
            const float* __restrict__ mx_b,
            float* __restrict__ u, float* __restrict__ z,
            _Float16* __restrict__ rr, float* __restrict__ hx) {
  __shared__ _Float16 sA[128][40];
  __shared__ _Float16 sB[128][40];
  int bm = blockIdx.y * 128, bn = blockIdx.x * 128;
  v8f acc[2][4] = {};
  gemm_tile<false>(mxh, D_, mxw16, MXN_, D_, bm, bn, sA, sB, acc);
  EPILOGUE_BEGIN
    float val = av + mx_b[c];
    if (c < D_)               u [(long)r * D_ + c]             = sigm_(val);
    else if (c < D_ + Z_)     z [(long)r * Z_ + (c - D_)]      = silu_(val);
    else if (c < D_ + Z_ + H_)rr[(long)r * H_ + (c - D_ - Z_)] = (_Float16)silu_(val);
    else                      hx[(long)r * D_ + (c - D_ - Z_ - H_)] = val;
  EPILOGUE_END
}

// ---------- batched scores: S = q @ k^T + rel_bias ----------
__global__ void __launch_bounds__(256)
k_scores(const _Float16* __restrict__ qh, const _Float16* __restrict__ kh,
         const float* __restrict__ rel_b, float* __restrict__ sc) {
  __shared__ _Float16 sA[128][40];
  __shared__ _Float16 sB[128][40];
  int beta = blockIdx.z;
  int b = beta >> 4, nch = beta & 15;
  long off = (long)nch * C_ * (B_ * Z_) + (long)b * Z_;
  int bm = blockIdx.y * 128, bn = blockIdx.x * 128;
  v8f acc[2][4] = {};
  gemm_tile<true>(qh + off, B_ * Z_, kh + off, B_ * Z_, Z_, bm, bn, sA, sB, acc);
  EPILOGUE_BEGIN
    sc[((long)beta * C_ + r) * C_ + c] = av + rel_b[C_ - 1 + c - r];
  EPILOGUE_END
}

// ---------- batched h = attn @ v, epilogue multiplies by r ----------
__global__ void __launch_bounds__(256)
k_hv(const _Float16* __restrict__ attn, const _Float16* __restrict__ vbuf,
     const _Float16* __restrict__ rr, _Float16* __restrict__ hr) {
  __shared__ _Float16 sA[128][40];
  __shared__ _Float16 sB[128][40];
  int beta = blockIdx.z;
  int b = beta >> 4, nch = beta & 15;
  const _Float16* A = attn + (long)beta * C_ * C_;
  const _Float16* Bv = vbuf + (long)nch * C_ * (B_ * H_) + (long)b * H_;
  int bm = blockIdx.y * 128, bn = blockIdx.x * 128;
  v8f acc[2][4] = {};
  gemm_tile<false>(A, C_, Bv, B_ * H_, C_, bm, bn, sA, sB, acc);
  EPILOGUE_BEGIN
    long grow = (long)(nch * C_ + r) * B_ + b;
    long idx  = grow * H_ + c;
    hr[idx] = (_Float16)(av * (float)rr[idx]);
  EPILOGUE_END
}

// ---------- final: out = x + u * (silu(hx + hr@h_w + h_b) - x) ----------
__global__ void __launch_bounds__(256)
k_final(const _Float16* __restrict__ hr, const _Float16* __restrict__ hw16,
        const float* __restrict__ h_b, const float* __restrict__ hx,
        const float* __restrict__ u, const float* __restrict__ x,
        float* __restrict__ out) {
  __shared__ _Float16 sA[128][40];
  __shared__ _Float16 sB[128][40];
  int bm = blockIdx.y * 128, bn = blockIdx.x * 128;
  v8f acc[2][4] = {};
  gemm_tile<false>(hr, H_, hw16, D_, H_, bm, bn, sA, sB, acc);
  EPILOGUE_BEGIN
    long idx = (long)r * D_ + c;
    float hval = silu_(hx[idx] + av + h_b[c]);
    float res  = x[idx];
    out[idx] = res + u[idx] * (hval - res);
  EPILOGUE_END
}

// ---------- LayerNorm: one block per row of 1024 ----------
__global__ void __launch_bounds__(256)
k_ln(const float* __restrict__ x, const float* __restrict__ w, const float* __restrict__ b,
     float* __restrict__ xn, _Float16* __restrict__ xnh) {
  long row = blockIdx.x;
  const float* xr = x + row * D_;
  float4 v = ((const float4*)xr)[threadIdx.x];
  float s  = v.x + v.y + v.z + v.w;
  float s2 = v.x*v.x + v.y*v.y + v.z*v.z + v.w*v.w;
  #pragma unroll
  for (int o = 16; o > 0; o >>= 1) { s += __shfl_xor(s, o, 32); s2 += __shfl_xor(s2, o, 32); }
  __shared__ float rs[8], rs2[8];
  int lane = threadIdx.x & 31, wave = threadIdx.x >> 5;
  if (lane == 0) { rs[wave] = s; rs2[wave] = s2; }
  __syncthreads();
  float S = 0.f, S2 = 0.f;
  #pragma unroll
  for (int i = 0; i < 8; ++i) { S += rs[i]; S2 += rs2[i]; }
  float mu  = S * (1.0f / D_);
  float var = S2 * (1.0f / D_) - mu * mu;
  float inv = rsqrtf(var + 1e-5f);
  float vv[4] = {v.x, v.y, v.z, v.w};
  #pragma unroll
  for (int q = 0; q < 4; ++q) {
    int c = threadIdx.x * 4 + q;
    float o = (vv[q] - mu) * inv * w[c] + b[c];
    xn [row * D_ + c] = o;
    xnh[row * D_ + c] = (_Float16)o;
  }
}

// ---------- EMA recurrence: one thread per (b,d), 16 damped states ----------
__global__ void __launch_bounds__(256)
k_ema(const float* __restrict__ xn,
      const float* __restrict__ e_delta, const float* __restrict__ e_alpha,
      const float* __restrict__ e_beta,  const float* __restrict__ e_gamma,
      const float* __restrict__ e_omega, _Float16* __restrict__ mxh) {
  int t = blockIdx.x * 256 + threadIdx.x;   // [0, 4096)
  int b = t >> 10, d = t & 1023;
  float q[NEMA], cc[NEMA], s[NEMA];
  #pragma unroll
  for (int n = 0; n < NEMA; ++n) {
    float p = sigm_(e_delta[d * NEMA + n]);
    q[n]  = 1.0f - p * sigm_(e_alpha[d * NEMA + n]);
    cc[n] = p * e_beta[d * NEMA + n] * e_gamma[d * NEMA + n] * 0.25f; // EMA_SCALE = 1/sqrt(16)
    s[n]  = 0.0f;
  }
  float om = e_omega[d];
  for (int l = 0; l < L_; ++l) {
    long idx = ((long)(l * B_ + b)) * D_ + d;
    float xv = xn[idx];
    float cv = 0.0f;
    #pragma unroll
    for (int n = 0; n < NEMA; ++n) { s[n] = q[n] * s[n] + xv; cv += cc[n] * s[n]; }
    mxh[idx] = (_Float16)silu_(cv + xv * om);
  }
}

// ---------- q/k prep: f16 operands with gamma/beta and Z^-0.5 ----------
__global__ void __launch_bounds__(256)
k_qk(const float* __restrict__ z, const float* __restrict__ gamma, const float* __restrict__ beta,
     _Float16* __restrict__ qh, _Float16* __restrict__ kh) {
  long i = (long)blockIdx.x * 256 + threadIdx.x;  // ROWS_*Z_
  int zi = (int)(i & (Z_ - 1));
  float zv = z[i];
  qh[i] = (_Float16)((zv * gamma[zi] + beta[zi]) * 0.08838834764831845f);
  kh[i] = (_Float16)(zv * gamma[Z_ + zi] + beta[Z_ + zi]);
}

// ---------- softmax: one wave per row of 256 ----------
__global__ void __launch_bounds__(256)
k_softmax(const float* __restrict__ sc, _Float16* __restrict__ attn) {
  int wave = threadIdx.x >> 5, lane = threadIdx.x & 31;
  long row = (long)blockIdx.x * 8 + wave;
  const float* p = sc + row * C_;
  float v[8], mx = -1e30f;
  #pragma unroll
  for (int i = 0; i < 8; ++i) { v[i] = p[lane + i * 32]; mx = fmaxf(mx, v[i]); }
  #pragma unroll
  for (int o = 16; o > 0; o >>= 1) mx = fmaxf(mx, __shfl_xor(mx, o, 32));
  float sum = 0.f;
  #pragma unroll
  for (int i = 0; i < 8; ++i) { v[i] = __expf(v[i] - mx); sum += v[i]; }
  #pragma unroll
  for (int o = 16; o > 0; o >>= 1) sum += __shfl_xor(sum, o, 32);
  float inv = __builtin_amdgcn_rcpf(sum);
  #pragma unroll
  for (int i = 0; i < 8; ++i) attn[row * C_ + lane + i * 32] = (_Float16)(v[i] * inv);
}

// ---------- f32 -> f16 convert ----------
__global__ void k_cvt(const float* __restrict__ in, _Float16* __restrict__ out, long n) {
  long i = (long)blockIdx.x * blockDim.x + threadIdx.x;
  long st = (long)gridDim.x * blockDim.x;
  for (; i < n; i += st) out[i] = (_Float16)in[i];
}

extern "C" void kernel_launch(void* const* d_in, const int* in_sizes, int n_in,
                              void* d_out, int out_size, void* d_ws, size_t ws_size,
                              hipStream_t stream) {
  (void)in_sizes; (void)n_in; (void)out_size; (void)ws_size;
  const float* x       = (const float*)d_in[0];
  const float* ln_w    = (const float*)d_in[1];
  const float* ln_b    = (const float*)d_in[2];
  const float* v_w     = (const float*)d_in[3];
  const float* v_b     = (const float*)d_in[4];
  const float* mx_w    = (const float*)d_in[5];
  const float* mx_b    = (const float*)d_in[6];
  const float* h_w     = (const float*)d_in[7];
  const float* h_b     = (const float*)d_in[8];
  const float* gamma   = (const float*)d_in[9];
  const float* beta    = (const float*)d_in[10];
  const float* rel_b   = (const float*)d_in[11];
  const float* e_delta = (const float*)d_in[12];
  const float* e_alpha = (const float*)d_in[13];
  const float* e_beta  = (const float*)d_in[14];
  const float* e_gamma = (const float*)d_in[15];
  const float* e_omega = (const float*)d_in[16];
  float* out = (float*)d_out;

  char* w = (char*)d_ws;
  size_t off = 0;
  auto alloc = [&](size_t bytes) -> void* {
    void* p = w + off; off += (bytes + 255) & ~(size_t)255; return p;
  };
  float*     xn    = (float*)    alloc((size_t)ROWS_ * D_ * 4);
  _Float16*  xnh   = (_Float16*) alloc((size_t)ROWS_ * D_ * 2);
  _Float16*  mxh   = (_Float16*) alloc((size_t)ROWS_ * D_ * 2);
  _Float16*  vw16  = (_Float16*) alloc((size_t)D_ * H_ * 2);
  _Float16*  mxw16 = (_Float16*) alloc((size_t)D_ * MXN_ * 2);
  _Float16*  hw16  = (_Float16*) alloc((size_t)H_ * D_ * 2);
  _Float16*  vbuf  = (_Float16*) alloc((size_t)ROWS_ * H_ * 2);
  float*     u     = (float*)    alloc((size_t)ROWS_ * D_ * 4);
  float*     zb    = (float*)    alloc((size_t)ROWS_ * Z_ * 4);
  _Float16*  rr    = (_Float16*) alloc((size_t)ROWS_ * H_ * 2);
  float*     hx    = (float*)    alloc((size_t)ROWS_ * D_ * 4);
  _Float16*  qh    = (_Float16*) alloc((size_t)ROWS_ * Z_ * 2);
  _Float16*  kh    = (_Float16*) alloc((size_t)ROWS_ * Z_ * 2);
  float*     sc    = (float*)    alloc((size_t)B_ * 16 * C_ * C_ * 4);
  _Float16*  attn  = (_Float16*) alloc((size_t)B_ * 16 * C_ * C_ * 2);
  _Float16*  hr    = (_Float16*) alloc((size_t)ROWS_ * H_ * 2);

  // weight conversions (independent)
  k_cvt<<<4096, 256, 0, stream>>>(v_w,  vw16,  (long)D_ * H_);
  k_cvt<<<4096, 256, 0, stream>>>(mx_w, mxw16, (long)D_ * MXN_);
  k_cvt<<<4096, 256, 0, stream>>>(h_w,  hw16,  (long)H_ * D_);

  // LayerNorm
  k_ln<<<ROWS_, 256, 0, stream>>>(x, ln_w, ln_b, xn, xnh);

  // EMA recurrence -> mx (f16)
  k_ema<<<16, 256, 0, stream>>>(xn, e_delta, e_alpha, e_beta, e_gamma, e_omega, mxh);

  // v = silu(xn @ v_w + v_b)
  k_gemm_v<<<dim3(H_ / 128, ROWS_ / 128), 256, 0, stream>>>(xnh, vw16, v_b, vbuf);

  // base = mx @ mx_w + mx_b  -> u, z, r, hx
  k_gemm_base<<<dim3(MXN_ / 128, ROWS_ / 128), 256, 0, stream>>>(mxh, mxw16, mx_b, u, zb, rr, hx);

  // q/k prep
  k_qk<<<(ROWS_ * Z_) / 256, 256, 0, stream>>>(zb, gamma, beta, qh, kh);

  // scores (64 batches of 256x256x128)
  k_scores<<<dim3(2, 2, 64), 256, 0, stream>>>(qh, kh, rel_b, sc);

  // softmax -> attn f16
  k_softmax<<<(B_ * 16 * C_) / 8, 256, 0, stream>>>(sc, attn);

  // h = attn @ v, times r  (64 batches of 256x2048x256)
  k_hv<<<dim3(H_ / 128, 2, 64), 256, 0, stream>>>(attn, vbuf, rr, hr);

  // out = x + u * (silu(hx + hr@h_w + h_b) - x)
  k_final<<<dim3(D_ / 128, ROWS_ / 128), 256, 0, stream>>>(hr, hw16, h_b, hx, u, x, out);
}
